// HomeoTopoBrain_18769007084208
// MI455X (gfx1250) — matrix operand order repaired and checked by reading, not compiled, not run on hardware
//
#include <hip/hip_runtime.h>
#include <math.h>

// CDNA5 (gfx1250) wave32 WMMA types
typedef __attribute__((ext_vector_type(16))) __bf16 v16bf;
typedef __attribute__((ext_vector_type(4)))  __bf16 v4bf;
typedef __attribute__((ext_vector_type(8)))  float  v8f;

#define BATCH 4096
#define NNODE 64
#define DDIM  64
#define NDTOT 4096   // NNODE*DDIM
#define NFEAT 256
#define NCLS  10

__device__ __forceinline__ float sigf(float x) { return 1.f / (1.f + expf(-x)); }

// 5 -> 16 -> 8 -> 3 regulator MLP with LayerNorms + tanh, sigmoid head (scalar per thread)
__device__ void regulator_mlp(const float sv[5],
    const float* __restrict__ W1, const float* __restrict__ b1,
    const float* __restrict__ g1, const float* __restrict__ bb1,
    const float* __restrict__ W2, const float* __restrict__ b2,
    const float* __restrict__ g2, const float* __restrict__ bb2,
    const float* __restrict__ W3, const float* __restrict__ b3,
    float out[3]) {
  float h1[16];
  #pragma unroll
  for (int i = 0; i < 16; ++i) {
    float a = b1[i];
    #pragma unroll
    for (int j = 0; j < 5; ++j) a += W1[i * 5 + j] * sv[j];
    h1[i] = a;
  }
  float mu = 0.f;
  #pragma unroll
  for (int i = 0; i < 16; ++i) mu += h1[i];
  mu *= (1.f / 16.f);
  float var = 0.f;
  #pragma unroll
  for (int i = 0; i < 16; ++i) { float d = h1[i] - mu; var += d * d; }
  var *= (1.f / 16.f);
  float inv = rsqrtf(var + 1e-5f);
  #pragma unroll
  for (int i = 0; i < 16; ++i) h1[i] = tanhf((h1[i] - mu) * inv * g1[i] + bb1[i]);

  float h2[8];
  #pragma unroll
  for (int i = 0; i < 8; ++i) {
    float a = b2[i];
    #pragma unroll
    for (int j = 0; j < 16; ++j) a += W2[i * 16 + j] * h1[j];
    h2[i] = a;
  }
  mu = 0.f;
  #pragma unroll
  for (int i = 0; i < 8; ++i) mu += h2[i];
  mu *= (1.f / 8.f);
  var = 0.f;
  #pragma unroll
  for (int i = 0; i < 8; ++i) { float d = h2[i] - mu; var += d * d; }
  var *= (1.f / 8.f);
  inv = rsqrtf(var + 1e-5f);
  #pragma unroll
  for (int i = 0; i < 8; ++i) h2[i] = tanhf((h2[i] - mu) * inv * g2[i] + bb2[i]);

  #pragma unroll
  for (int i = 0; i < 3; ++i) {
    float a = b3[i];
    #pragma unroll
    for (int j = 0; j < 8; ++j) a += W3[i * 8 + j] * h2[j];
    out[i] = sigf(a);
  }
}

// ------------------------------------------------------------------
// Kernel 1: e = x(4096x256) @ W_embed^T(256x4096) + b_embed, via bf16 WMMA.
// grid (64,64) of 64x64 tiles, 256 threads = 8 waves, 2 C-tiles per wave.
// Double-buffered LDS staging with b128 loads + packed bf16 b64 stores.
// ------------------------------------------------------------------
__global__ __launch_bounds__(256) void k_embed(
    const float* __restrict__ x, const float* __restrict__ We,
    const float* __restrict__ be, float* __restrict__ e) {
  __shared__ __bf16 As[2][64][36];   // [buf][m][k] chunk, bf16
  __shared__ __bf16 Bs[2][64][36];   // [buf][n][k] chunk (W_embed rows), bf16
  const int lane = threadIdx.x & 31;
  const int wave = threadIdx.x >> 5;
  const int h = lane >> 4;        // wave32 half
  const int l15 = lane & 15;
  const int tM = blockIdx.x * 64, tN = blockIdx.y * 64;

  // stage one 64x32 chunk of both matrices into LDS buffer `buf`
  auto stage = [&](int k0, int buf) {
    for (int i = threadIdx.x; i < 64 * 8; i += 256) {   // 512 float4-quads per matrix
      int r = i >> 3, q = (i & 7) * 4;
      float4 xa = *(const float4*)&x[(size_t)(tM + r) * NFEAT + k0 + q];
      float4 xb = *(const float4*)&We[(size_t)(tN + r) * NFEAT + k0 + q];
      v4bf a4 = { (__bf16)xa.x, (__bf16)xa.y, (__bf16)xa.z, (__bf16)xa.w };
      v4bf b4 = { (__bf16)xb.x, (__bf16)xb.y, (__bf16)xb.z, (__bf16)xb.w };
      *(v4bf*)&As[buf][r][q] = a4;
      *(v4bf*)&Bs[buf][r][q] = b4;
    }
  };

  v8f zero = {0.f, 0.f, 0.f, 0.f, 0.f, 0.f, 0.f, 0.f};
  v8f acc[2];
  acc[0] = zero; acc[1] = zero;

  stage(0, 0);
  for (int kc = 0; kc < 8; ++kc) {
    __syncthreads();
    if (kc < 7) stage((kc + 1) * 32, (kc + 1) & 1);   // overlap with WMMA below
    const int buf = kc & 1;
    #pragma unroll
    for (int it = 0; it < 2; ++it) {
      int t = wave + it * 8;               // 16 C tiles, 4x4; both tiles share tn
      int tm = (t >> 2) * 16, tn = (t & 3) * 16;
      v16bf a, b;
      #pragma unroll
      for (int v = 0; v < 8; ++v) {
        // A 16x32 bf16 layout: lanes0-15: K 0-7 (V0-3), 16-23 (V4-7); lanes16-31: +8
        int ka = (v < 4) ? (v * 2 + h * 8) : (16 + (v - 4) * 2 + h * 8);
        a[2 * v]     = As[buf][tm + l15][ka];
        a[2 * v + 1] = As[buf][tm + l15][ka + 1];
        // B 32x16 bf16: same K layout as A (lanes0-15: K 0-7, 16-23; lanes16-31: +8)
        int kb = (v < 4) ? (v * 2 + h * 8) : (16 + (v - 4) * 2 + h * 8);
        b[2 * v]     = Bs[buf][tn + l15][kb];
        b[2 * v + 1] = Bs[buf][tn + l15][kb + 1];
      }
      acc[it] = __builtin_amdgcn_wmma_f32_16x16x32_bf16(
          false, a, false, b, (short)0, acc[it], false, false);
    }
  }
  #pragma unroll
  for (int it = 0; it < 2; ++it) {
    int t = wave + it * 8;
    int tm = (t >> 2) * 16, tn = (t & 3) * 16;
    int n = tN + tn + l15;
    float bias = be[n];
    #pragma unroll
    for (int r = 0; r < 8; ++r) {
      int m = tM + tm + h * 8 + r;          // C layout: M = 8*half + vgpr
      e[(size_t)m * NDTOT + n] = acc[it][r] + bias;
    }
  }
}

// ------------------------------------------------------------------
// Kernel 2: per-block partial stats of e (sum, sumsq, sum|.|, min, max)
// ------------------------------------------------------------------
__global__ __launch_bounds__(256) void k_reduce_e(
    const float* __restrict__ e, float* __restrict__ part) {
  __shared__ float s0[256], s1[256], s2[256], s3[256], s4[256];
  int tid = threadIdx.x;
  const float4* e4 = (const float4*)(e + (size_t)blockIdx.x * 4096);
  float sm = 0.f, sq = 0.f, sa = 0.f, mn = 3.0e38f, mx = -3.0e38f;
  #pragma unroll
  for (int j = 0; j < 4; ++j) {
    float4 v4 = e4[tid + 256 * j];
    float vv[4] = {v4.x, v4.y, v4.z, v4.w};
    #pragma unroll
    for (int u = 0; u < 4; ++u) {
      float v = vv[u];
      sm += v; sq += v * v; sa += fabsf(v);
      mn = fminf(mn, v); mx = fmaxf(mx, v);
    }
  }
  s0[tid] = sm; s1[tid] = sq; s2[tid] = sa; s3[tid] = mn; s4[tid] = mx;
  __syncthreads();
  for (int st = 128; st > 0; st >>= 1) {
    if (tid < st) {
      s0[tid] += s0[tid + st];
      s1[tid] += s1[tid + st];
      s2[tid] += s2[tid + st];
      s3[tid] = fminf(s3[tid], s3[tid + st]);
      s4[tid] = fmaxf(s4[tid], s4[tid + st]);
    }
    __syncthreads();
  }
  if (tid == 0) {
    float* p = part + (size_t)blockIdx.x * 8;
    p[0] = s0[0]; p[1] = s1[0]; p[2] = s2[0]; p[3] = s3[0]; p[4] = s4[0];
  }
}

// ------------------------------------------------------------------
// Kernel 3: finish stats -> top regulator -> ap; build normalized sparse adj;
// also ||W_slow||_F for the cell regulator.
// ------------------------------------------------------------------
__global__ __launch_bounds__(256) void k_finalize(
    const float* __restrict__ part,
    const float* __restrict__ adjw, const float* __restrict__ Wslow,
    const float* __restrict__ tW1, const float* __restrict__ tb1,
    const float* __restrict__ tg1, const float* __restrict__ tbb1,
    const float* __restrict__ tW2, const float* __restrict__ tb2,
    const float* __restrict__ tg2, const float* __restrict__ tbb2,
    const float* __restrict__ tW3, const float* __restrict__ tb3,
    float* __restrict__ adjn, float* __restrict__ scal) {
  __shared__ float s0[256], s1[256], s2[256], s3[256], s4[256], s5[256], s6[256];
  __shared__ float s_ap;
  int tid = threadIdx.x;
  float sm = 0.f, sq = 0.f, sa = 0.f, mn = 3.0e38f, mx = -3.0e38f, aw = 0.f, wsq = 0.f;
  for (int j = tid; j < 4096; j += 256) {
    const float* p = part + (size_t)j * 8;
    sm += p[0]; sq += p[1]; sa += p[2];
    mn = fminf(mn, p[3]); mx = fmaxf(mx, p[4]);
    float a = adjw[j]; aw += a * a;
    float w = Wslow[j]; wsq += w * w;
  }
  s0[tid] = sm; s1[tid] = sq; s2[tid] = sa; s3[tid] = mn; s4[tid] = mx;
  s5[tid] = aw; s6[tid] = wsq;
  __syncthreads();
  for (int st = 128; st > 0; st >>= 1) {
    if (tid < st) {
      s0[tid] += s0[tid + st];
      s1[tid] += s1[tid + st];
      s2[tid] += s2[tid + st];
      s3[tid] = fminf(s3[tid], s3[tid + st]);
      s4[tid] = fmaxf(s4[tid], s4[tid + st]);
      s5[tid] += s5[tid + st];
      s6[tid] += s6[tid + st];
    }
    __syncthreads();
  }
  if (tid == 0) {
    const float n = 16777216.f;
    float mean = s0[0] / n;
    float var = (s1[0] - s0[0] * mean) / (n - 1.f);   // ddof=1
    float natural = fabsf(var - 0.5f);
    float smooth = (s4[0] - s3[0]) / (n - 1.f);       // sorted-diff mean telescopes
    float adv = fabsf(smooth - 1.f);
    float excit = s2[0] / n;
    float fat = sqrtf(s5[0]);                         // ||adj_weights||_F
    float gp = sqrtf(s1[0] / n);
    float sv[5] = {natural, adv, excit, fat, gp};
    float ctl[3];
    regulator_mlp(sv, tW1, tb1, tg1, tbb1, tW2, tb2, tg2, tbb2, tW3, tb3, ctl);
    scal[0] = ctl[0];                                 // ap
    scal[1] = sqrtf(s6[0]);                           // ||W_slow||_F
    s_ap = ctl[0];
  }
  __syncthreads();
  for (int i = tid; i < 4096; i += 256) adjn[i] = 0.f;
  __syncthreads();
  if (tid < 64) {
    float ap = s_ap;
    int r = tid >> 3, c = tid & 7;
    int nb[4]; float v[4]; int cnt = 0;
    if (r > 0) nb[cnt++] = tid - 8;
    if (r < 7) nb[cnt++] = tid + 8;
    if (c > 0) nb[cnt++] = tid - 1;
    if (c < 7) nb[cnt++] = tid + 1;
    float rsum = 0.f;
    for (int k = 0; k < cnt; ++k) { v[k] = sigf(adjw[tid * 64 + nb[k]] * ap); rsum += v[k]; }
    float inv = 1.f / fmaxf(rsum, 1e-6f);
    for (int k = 0; k < cnt; ++k) adjn[tid * 64 + nb[k]] = v[k] * inv;
  }
}

// ------------------------------------------------------------------
// Kernel 4: one block per batch element. message pass -> WMMA (W_slow,W_fast)
// -> per-row regulator -> swish(beta)+LN -> fused 4096->10 readout.
// Static LDS ~50KB (eS reused as proc buffer).
// ------------------------------------------------------------------
__global__ __launch_bounds__(256) void k_cell(
    const float* __restrict__ e, const float* __restrict__ adjn,
    const float* __restrict__ scal,
    const float* __restrict__ Wslow, const float* __restrict__ Wfast,
    const float* __restrict__ lng, const float* __restrict__ lnb,
    const float* __restrict__ Wread, const float* __restrict__ bread,
    const float* __restrict__ cW1, const float* __restrict__ cb1,
    const float* __restrict__ cg1, const float* __restrict__ cbb1,
    const float* __restrict__ cW2, const float* __restrict__ cb2,
    const float* __restrict__ cg2, const float* __restrict__ cbb2,
    const float* __restrict__ cW3, const float* __restrict__ cb3,
    float* __restrict__ out) {
  __shared__ float eS[4096];   // e tile, later reused as proc
  __shared__ float mS[4096];   // message output
  __shared__ float hS[4096];   // h_raw, later comb
  __shared__ float betaS[64], gateS[64];
  __shared__ float red[256];
  const int tid = threadIdx.x;
  const int lane = tid & 31, wave = tid >> 5;
  const int h = lane >> 4, l15 = lane & 15;
  const int b = blockIdx.x;
  const float* eb = e + (size_t)b * NDTOT;

  // warm the L2-resident readout panel (gfx1250 global_prefetch path)
  for (int i = tid * 64; i < NCLS * NDTOT; i += 256 * 64)
    __builtin_prefetch(&Wread[i], 0, 1);

  const float4* eb4 = (const float4*)eb;
  for (int i = tid; i < 1024; i += 256) {
    float4 v4 = eb4[i];
    *(float4*)&eS[i * 4] = v4;           // ds_store_b128
  }
  __syncthreads();

  // sparse message pass: m[n][d] = sum_nb adj[n][nb]*e[nb][d] (<=4 neighbors)
  for (int i = tid; i < 4096; i += 256) {
    int n = i >> 6, d = i & 63;
    int r = n >> 3, c = n & 7;
    float acc = 0.f;
    if (r > 0) acc += adjn[n * 64 + n - 8] * eS[(n - 8) * 64 + d];
    if (r < 7) acc += adjn[n * 64 + n + 8] * eS[(n + 8) * 64 + d];
    if (c > 0) acc += adjn[n * 64 + n - 1] * eS[(n - 1) * 64 + d];
    if (c < 7) acc += adjn[n * 64 + n + 1] * eS[(n + 1) * 64 + d];
    mS[i] = acc;
  }
  __syncthreads();

  // WMMA: hS = m @ Wslow^T (to LDS); accF = m @ Wfast^T (kept in VGPRs)
  v8f zero = {0.f, 0.f, 0.f, 0.f, 0.f, 0.f, 0.f, 0.f};
  v8f accF[2];
  #pragma unroll
  for (int it = 0; it < 2; ++it) {
    int t = wave + it * 8;
    int tm = (t >> 2) * 16, tn = (t & 3) * 16;
    v8f accS = zero;
    accF[it] = zero;
    #pragma unroll
    for (int k0 = 0; k0 < 64; k0 += 32) {
      v16bf a, bs, bf;
      #pragma unroll
      for (int v = 0; v < 8; ++v) {
        int ka = k0 + ((v < 4) ? (v * 2 + h * 8) : (16 + (v - 4) * 2 + h * 8));
        a[2 * v]     = (__bf16)mS[(tm + l15) * 64 + ka];
        a[2 * v + 1] = (__bf16)mS[(tm + l15) * 64 + ka + 1];
        int kb = k0 + ((v < 4) ? (v * 2 + h * 8) : (16 + (v - 4) * 2 + h * 8));   // B[k][n] = W[n][k] -> read W row-major
        bs[2 * v]     = (__bf16)Wslow[(tn + l15) * 64 + kb];
        bs[2 * v + 1] = (__bf16)Wslow[(tn + l15) * 64 + kb + 1];
        bf[2 * v]     = (__bf16)Wfast[(tn + l15) * 64 + kb];
        bf[2 * v + 1] = (__bf16)Wfast[(tn + l15) * 64 + kb + 1];
      }
      accS = __builtin_amdgcn_wmma_f32_16x16x32_bf16(
          false, a, false, bs, (short)0, accS, false, false);
      accF[it] = __builtin_amdgcn_wmma_f32_16x16x32_bf16(
          false, a, false, bf, (short)0, accF[it], false, false);
    }
    #pragma unroll
    for (int r = 0; r < 8; ++r)
      hS[(tm + h * 8 + r) * 64 + tn + l15] = accS[r];
  }
  __syncthreads();

  // per-row regulator (row = cell row, 64 rows)
  if (tid < 64) {
    float sm = 0.f, sq = 0.f, mn = 3.0e38f, mx = -3.0e38f, sa = 0.f, sh = 0.f;
    const float* mr = mS + tid * 64;
    const float* hr = hS + tid * 64;
    for (int d = 0; d < 64; ++d) {
      float v = mr[d]; sm += v; sq += v * v;
      mn = fminf(mn, v); mx = fmaxf(mx, v);
      float hv = hr[d]; sa += fabsf(hv); sh += hv * hv;
    }
    float var = (sq - sm * sm * (1.f / 64.f)) * (1.f / 63.f);  // ddof=1
    float sv[5] = { fabsf(var - 0.5f),
                    fabsf((mx - mn) * (1.f / 63.f) - 1.f),
                    sa * (1.f / 64.f),
                    scal[1],
                    sqrtf(sh * (1.f / 64.f)) };
    float ctl[3];
    regulator_mlp(sv, cW1, cb1, cg1, cbb1, cW2, cb2, cg2, cbb2, cW3, cb3, ctl);
    betaS[tid] = 0.5f + ctl[1] * 2.f;
    gateS[tid] = ctl[2];
  }
  __syncthreads();

  // comb = h_slow + h_fast*gate, written back into hS
  #pragma unroll
  for (int it = 0; it < 2; ++it) {
    int t = wave + it * 8;
    int tm = (t >> 2) * 16, tn = (t & 3) * 16;
    #pragma unroll
    for (int r = 0; r < 8; ++r) {
      int m = tm + h * 8 + r;
      int idx = m * 64 + tn + l15;
      hS[idx] = hS[idx] + accF[it][r] * gateS[m];
    }
  }
  __syncthreads();

  // swish(beta) + LayerNorm per row -> eS (proc)
  if (tid < 64) {
    float beta = betaS[tid];
    const float* cr = hS + tid * 64;
    float s1v = 0.f, s2v = 0.f;
    for (int d = 0; d < 64; ++d) {
      float comb = cr[d];
      float a = comb * sigf(beta * comb);
      s1v += a; s2v += a * a;
    }
    float mu = s1v * (1.f / 64.f);
    float var = fmaxf(s2v * (1.f / 64.f) - mu * mu, 0.f);
    float inv = rsqrtf(var + 1e-5f);
    float* pr = eS + tid * 64;
    for (int d = 0; d < 64; ++d) {
      float comb = cr[d];
      float a = comb * sigf(beta * comb);
      pr[d] = (a - mu) * inv * lng[d] + lnb[d];
    }
  }
  __syncthreads();

  // fused readout: out[b][c] = proc_row(4096) . W_read[c] + b_read[c]
  float acc[NCLS];
  #pragma unroll
  for (int c = 0; c < NCLS; ++c) acc[c] = 0.f;
  for (int j = 0; j < 16; ++j) {
    int i = tid + 256 * j;
    float p = eS[i];
    #pragma unroll
    for (int c = 0; c < NCLS; ++c) acc[c] += p * Wread[(size_t)c * NDTOT + i];
  }
  for (int c = 0; c < NCLS; ++c) {
    red[tid] = acc[c];
    __syncthreads();
    for (int st = 128; st > 0; st >>= 1) {
      if (tid < st) red[tid] += red[tid + st];
      __syncthreads();
    }
    if (tid == 0) out[(size_t)b * NCLS + c] = red[0] + bread[c];
    __syncthreads();
  }
}

// ------------------------------------------------------------------
extern "C" void kernel_launch(void* const* d_in, const int* in_sizes, int n_in,
                              void* d_out, int out_size, void* d_ws, size_t ws_size,
                              hipStream_t stream) {
  (void)in_sizes; (void)n_in; (void)out_size; (void)ws_size;
  const float* x       = (const float*)d_in[0];
  const float* W_embed = (const float*)d_in[1];
  const float* b_embed = (const float*)d_in[2];
  const float* adj_w   = (const float*)d_in[3];
  const float* W_slow  = (const float*)d_in[4];
  const float* W_fast  = (const float*)d_in[5];
  const float* ln_g    = (const float*)d_in[6];
  const float* ln_b    = (const float*)d_in[7];
  const float* W_read  = (const float*)d_in[8];
  const float* b_read  = (const float*)d_in[9];
  // t params: indices 10..19, c params: 20..29
  const float* t0 = (const float*)d_in[10]; const float* t1 = (const float*)d_in[11];
  const float* t2 = (const float*)d_in[12]; const float* t3 = (const float*)d_in[13];
  const float* t4 = (const float*)d_in[14]; const float* t5 = (const float*)d_in[15];
  const float* t6 = (const float*)d_in[16]; const float* t7 = (const float*)d_in[17];
  const float* t8 = (const float*)d_in[18]; const float* t9 = (const float*)d_in[19];
  const float* c0 = (const float*)d_in[20]; const float* c1 = (const float*)d_in[21];
  const float* c2 = (const float*)d_in[22]; const float* c3 = (const float*)d_in[23];
  const float* c4 = (const float*)d_in[24]; const float* c5 = (const float*)d_in[25];
  const float* c6 = (const float*)d_in[26]; const float* c7 = (const float*)d_in[27];
  const float* c8 = (const float*)d_in[28]; const float* c9 = (const float*)d_in[29];

  float* ws   = (float*)d_ws;
  float* e    = ws;                       // 4096*4096 = 16,777,216 floats (64MB)
  float* part = ws + 16777216;            // 4096 * 8 = 32,768 floats
  float* adjn = part + 32768;             // 64*64 = 4096 floats
  float* scal = adjn + 4096;              // [ap, ||W_slow||]

  k_embed<<<dim3(64, 64), 256, 0, stream>>>(x, W_embed, b_embed, e);
  k_reduce_e<<<4096, 256, 0, stream>>>(e, part);
  k_finalize<<<1, 256, 0, stream>>>(part, adj_w, W_slow,
                                    t0, t1, t2, t3, t4, t5, t6, t7, t8, t9,
                                    adjn, scal);
  k_cell<<<4096, 256, 0, stream>>>(e, adjn, scal, W_slow, W_fast, ln_g, ln_b,
                                   W_read, b_read,
                                   c0, c1, c2, c3, c4, c5, c6, c7, c8, c9,
                                   (float*)d_out);
}
